// DGI_24489903521945
// MI455X (gfx1250) — matrix operand order repaired and checked by reading, not compiled, not run on hardware
//
#include <hip/hip_runtime.h>
#include <hip/hip_bf16.h>

// Problem constants (match reference)
#define NNODES 50000
#define NEDGES 800000
#define FDIM   256
#define HDIM   256
#define NPB    196   // ceil(NNODES/256) partial blocks for column mean

typedef __attribute__((ext_vector_type(16))) __bf16 v16bf;
typedef __attribute__((ext_vector_type(8)))  float  v8f;

static __device__ __forceinline__ __bf16 f2bf(float f) {
  union { float f; unsigned u; } x;
  x.f = f;
  unsigned r = (x.u + 0x7FFFu + ((x.u >> 16) & 1u)) >> 16;  // RNE
  unsigned short h = (unsigned short)r;
  __bf16 b;
  __builtin_memcpy(&b, &h, 2);
  return b;
}

// ---------------- utility kernels ----------------

__global__ void k_zero4(float4* __restrict__ p, long n4) {
  long i = (long)blockIdx.x * blockDim.x + threadIdx.x;
  long stride = (long)gridDim.x * blockDim.x;
  float4 z; z.x = z.y = z.z = z.w = 0.f;
  for (; i < n4; i += stride) p[i] = z;
}

__global__ void k_initdeg(int* __restrict__ deg) {
  int i = blockIdx.x * blockDim.x + threadIdx.x;
  if (i < NNODES) deg[i] = 1;  // self-loop
}

__global__ void k_deg(const int* __restrict__ ei, int* __restrict__ deg) {
  int e = blockIdx.x * blockDim.x + threadIdx.x;
  if (e < NEDGES) atomicAdd(&deg[ei[NEDGES + e]], 1);  // in-degree on dst
}

__global__ void k_dinv(const int* __restrict__ deg, float* __restrict__ dinv) {
  int i = blockIdx.x * blockDim.x + threadIdx.x;
  if (i < NNODES) dinv[i] = rsqrtf((float)deg[i]);
}

// ---------------- fragment-major packing (f32 -> bf16 once) ----------------
// Apack element index o = ((rt*8 + kt)*32 + lane)*16 + j
//   lane<16: row = rt*16+lane,    K = kt*32 + j        (K 0..15)
//   lane>=16: row = rt*16+lane-16, K = kt*32 + 16 + j   (K 16..31)
__global__ __launch_bounds__(256) void k_packA(const float* __restrict__ x,
                                               __bf16* __restrict__ Apack) {
  long o = (long)blockIdx.x * blockDim.x + threadIdx.x;
  if (o >= (long)NNODES * FDIM) return;
  int  j    = (int)(o & 15);
  int  lane = (int)((o >> 4) & 31);
  int  kt   = (int)((o >> 9) & 7);
  long rt   = o >> 12;
  int  half = lane >> 4, l16 = lane & 15;
  int  m = (int)(rt * 16 + l16);
  int  k = kt * 32 + (half ? 16 : 0) + j;
  Apack[o] = f2bf(x[(size_t)m * FDIM + k]);
}

// Bpack element index o = ((kt*16 + ct)*32 + lane)*16 + j
//   lane<16:  col = ct*16+lane,    K = kt*32 + j        (K 0..15)
//   lane>=16: col = ct*16+lane-16, K = kt*32 + 16 + j   (K 16..31)
__global__ __launch_bounds__(256) void k_packB(const float* __restrict__ W,
                                               __bf16* __restrict__ Bpack) {
  int o = blockIdx.x * blockDim.x + threadIdx.x;
  if (o >= FDIM * HDIM) return;
  int j    = o & 15;
  int lane = (o >> 4) & 31;
  int ct   = (o >> 9) & 15;
  int kt   = o >> 13;
  int half = lane >> 4, l16 = lane & 15;
  int n = ct * 16 + l16;
  int k = kt * 32 + (half ? 16 : 0) + j;
  Bpack[o] = f2bf(W[(size_t)k * HDIM + n]);
}

// ---------------- GEMM: xw = x @ W  (bf16 WMMA, f32 accumulate) ----------------
// grid = NNODES/16 blocks of 256 threads (8 waves). Block owns a 16-row strip,
// wave w owns columns [32w, 32w+32). Fragments are pre-packed: each lane loads
// its 16 bf16 elements as one aligned 32-byte vector (2x global_load_b128).
__global__ __launch_bounds__(256) void k_gemm(const v16bf* __restrict__ Apack,
                                              const v16bf* __restrict__ Bpack,
                                              float* __restrict__ xw) {
  const int lane = threadIdx.x & 31;
  const int wave = threadIdx.x >> 5;
  const int rt = blockIdx.x;

  v8f acc0 = {};
  v8f acc1 = {};

  const v16bf* Ap = Apack + (size_t)rt * 8 * 32 + lane;       // + kt*32
  const v16bf* Bp = Bpack + (size_t)(wave * 2) * 32 + lane;   // + kt*512, +32 for ct+1

#pragma unroll
  for (int kt = 0; kt < 8; ++kt) {
    v16bf a  = Ap[(size_t)kt * 32];
    v16bf b0 = Bp[(size_t)kt * 512];
    v16bf b1 = Bp[(size_t)kt * 512 + 32];
    acc0 = __builtin_amdgcn_wmma_f32_16x16x32_bf16(false, a, false, b0,
                                                   (short)0, acc0, false, false);
    acc1 = __builtin_amdgcn_wmma_f32_16x16x32_bf16(false, a, false, b1,
                                                   (short)0, acc1, false, false);
  }

  // C/D layout: VGPR i -> (M = rt*16 + 8*half + i, N = wave*32 + l16 [+16])
  const int half = lane >> 4, l16 = lane & 15;
  const int cm = rt * 16 + half * 8;
  const int cn = wave * 32 + l16;
#pragma unroll
  for (int i = 0; i < 8; ++i) {
    xw[(size_t)(cm + i) * HDIM + cn]      = acc0[i];
    xw[(size_t)(cm + i) * HDIM + cn + 16] = acc1[i];
  }
}

// ---------------- edge scatter: agg{1,2}[dst] += xw[{src,perm[src]}] * norm ----------
// 4 edges per block, 64 threads/edge, 4 floats/thread. xw (51 MB) is L2-resident.
__global__ __launch_bounds__(256) void k_scatter(const float* __restrict__ xw,
                                                 const int* __restrict__ ei,
                                                 const int* __restrict__ perm,
                                                 const float* __restrict__ dinv,
                                                 float* __restrict__ agg1,
                                                 float* __restrict__ agg2) {
  const int le = threadIdx.x >> 6;
  const int h  = (threadIdx.x & 63) * 4;
  const int e  = blockIdx.x * 4 + le;
  if (e >= NEDGES) return;
  const int s = ei[e];
  const int d = ei[NEDGES + e];
  const int p = perm[s];                       // corrupted branch: xw[perm[src]]
  const float norm = dinv[s] * dinv[d];
  const float4 v1 = *(const float4*)(xw + (size_t)s * HDIM + h);
  const float4 v2 = *(const float4*)(xw + (size_t)p * HDIM + h);
  float* o1 = agg1 + (size_t)d * HDIM + h;
  float* o2 = agg2 + (size_t)d * HDIM + h;
  atomicAdd(o1 + 0, v1.x * norm); atomicAdd(o1 + 1, v1.y * norm);
  atomicAdd(o1 + 2, v1.z * norm); atomicAdd(o1 + 3, v1.w * norm);
  atomicAdd(o2 + 0, v2.x * norm); atomicAdd(o2 + 1, v2.y * norm);
  atomicAdd(o2 + 2, v2.z * norm); atomicAdd(o2 + 3, v2.w * norm);
}

// ---------------- finalize: self-loop + bias + PReLU, in place ----------------
__global__ __launch_bounds__(256) void k_finalize(const float* __restrict__ xw,
                                                  const int* __restrict__ perm,
                                                  const float* __restrict__ dinv,
                                                  const float* __restrict__ bias,
                                                  const float* __restrict__ slope,
                                                  float* __restrict__ agg1,
                                                  float* __restrict__ agg2) {
  const int i = blockIdx.x;
  const int h = threadIdx.x;
  const float dii = dinv[i] * dinv[i];
  const int p = perm[i];
  const float al = slope[0];
  const float bb = bias[h];
  const size_t idx = (size_t)i * HDIM + h;
  float z1 = agg1[idx] + xw[idx] * dii + bb;
  float z2 = agg2[idx] + xw[(size_t)p * HDIM + h] * dii + bb;
  agg1[idx] = (z1 >= 0.f) ? z1 : al * z1;
  agg2[idx] = (z2 >= 0.f) ? z2 : al * z2;
}

// ---------------- column partial sums of z1 (coalesced) ----------------
__global__ __launch_bounds__(256) void k_colpart(const float* __restrict__ z1,
                                                 float* __restrict__ partial) {
  const int h = threadIdx.x;
  const int b = blockIdx.x;
  int i0 = b * 256;
  int i1 = i0 + 256; if (i1 > NNODES) i1 = NNODES;
  float s = 0.f;
  for (int i = i0; i < i1; ++i) s += z1[(size_t)i * HDIM + h];
  partial[(size_t)b * HDIM + h] = s;
}

// ---------------- summary = sigmoid(mean); wsum = disc_W @ summary -------------
__global__ __launch_bounds__(256) void k_wsum(const float* __restrict__ partial,
                                              const float* __restrict__ disc_W,
                                              float* __restrict__ wsum) {
  __shared__ float s[HDIM];
  const int h = threadIdx.x;
  float c = 0.f;
  for (int b = 0; b < NPB; ++b) c += partial[(size_t)b * HDIM + h];
  float mean = c * (1.0f / (float)NNODES);
  s[h] = 1.0f / (1.0f + expf(-mean));
  __syncthreads();
  float w = 0.f;
  for (int k = 0; k < HDIM; ++k) w += disc_W[(size_t)h * HDIM + k] * s[k];
  wsum[h] = w;
}

// ---------------- pos/neg = z{1,2} @ wsum, one wave per node -------------------
__global__ __launch_bounds__(256) void k_posneg(const float* __restrict__ z1,
                                                const float* __restrict__ z2,
                                                const float* __restrict__ wsum,
                                                float* __restrict__ out) {
  __shared__ float sw[HDIM];
  sw[threadIdx.x] = wsum[threadIdx.x];
  __syncthreads();
  const int lane = threadIdx.x & 31;
  const int wave = threadIdx.x >> 5;
  const int node = blockIdx.x * 8 + wave;
  if (node >= NNODES) return;
  const size_t base = (size_t)node * HDIM + lane * 8;
  float4 p0 = *(const float4*)(z1 + base);
  float4 p1 = *(const float4*)(z1 + base + 4);
  float4 q0 = *(const float4*)(z2 + base);
  float4 q1 = *(const float4*)(z2 + base + 4);
  const float* w = sw + lane * 8;
  float pos = p0.x*w[0] + p0.y*w[1] + p0.z*w[2] + p0.w*w[3]
            + p1.x*w[4] + p1.y*w[5] + p1.z*w[6] + p1.w*w[7];
  float neg = q0.x*w[0] + q0.y*w[1] + q0.z*w[2] + q0.w*w[3]
            + q1.x*w[4] + q1.y*w[5] + q1.z*w[6] + q1.w*w[7];
#pragma unroll
  for (int off = 16; off > 0; off >>= 1) {
    pos += __shfl_down(pos, off, 32);
    neg += __shfl_down(neg, off, 32);
  }
  if (lane == 0) { out[node] = pos; out[NNODES + node] = neg; }
}

static inline size_t alignup(size_t x, size_t a) { return (x + a - 1) & ~(a - 1); }

extern "C" void kernel_launch(void* const* d_in, const int* in_sizes, int n_in,
                              void* d_out, int out_size, void* d_ws, size_t ws_size,
                              hipStream_t stream) {
  const float* x      = (const float*)d_in[0];
  const float* W      = (const float*)d_in[1];
  const float* b      = (const float*)d_in[2];
  const float* a      = (const float*)d_in[3];
  const float* disc_W = (const float*)d_in[4];
  const int* edge_index = (const int*)d_in[5];
  const int* perm       = (const int*)d_in[6];
  float* out = (float*)d_out;

  // workspace layout (~180 MB), all 256-B aligned
  unsigned char* base = (unsigned char*)d_ws;
  size_t off = 0;
  float* xw      = (float*)(base + off);  off = alignup(off + (size_t)NNODES * HDIM * 4, 256);
  float* agg1    = (float*)(base + off);  off = alignup(off + (size_t)NNODES * HDIM * 4, 256);
  float* agg2    = (float*)(base + off);  off = alignup(off + (size_t)NNODES * HDIM * 4, 256);
  __bf16* Apack  = (__bf16*)(base + off); off = alignup(off + (size_t)NNODES * FDIM * 2, 256);
  __bf16* Bpack  = (__bf16*)(base + off); off = alignup(off + (size_t)FDIM * HDIM * 2, 256);
  int*   deg     = (int*)(base + off);    off = alignup(off + (size_t)NNODES * 4, 256);
  float* dinv    = (float*)(base + off);  off = alignup(off + (size_t)NNODES * 4, 256);
  float* partial = (float*)(base + off);  off = alignup(off + (size_t)NPB * HDIM * 4, 256);
  float* wsum    = (float*)(base + off);  off = alignup(off + (size_t)HDIM * 4, 256);

  // zero agg1+agg2 (each zeroed by its own launch)
  k_zero4<<<2048, 256, 0, stream>>>((float4*)agg1, (long)NNODES * HDIM / 4);
  k_zero4<<<2048, 256, 0, stream>>>((float4*)agg2, (long)NNODES * HDIM / 4);
  k_initdeg<<<(NNODES + 255) / 256, 256, 0, stream>>>(deg);
  k_deg<<<(NEDGES + 255) / 256, 256, 0, stream>>>(edge_index, deg);
  k_dinv<<<(NNODES + 255) / 256, 256, 0, stream>>>(deg, dinv);
  k_packA<<<(NNODES * FDIM + 255) / 256, 256, 0, stream>>>(x, Apack);
  k_packB<<<(FDIM * HDIM + 255) / 256, 256, 0, stream>>>(W, Bpack);
  k_gemm<<<NNODES / 16, 256, 0, stream>>>((const v16bf*)Apack, (const v16bf*)Bpack, xw);
  k_scatter<<<NEDGES / 4, 256, 0, stream>>>(xw, edge_index, perm, dinv, agg1, agg2);
  k_finalize<<<NNODES, 256, 0, stream>>>(xw, perm, dinv, b, a, agg1, agg2);
  k_colpart<<<NPB, 256, 0, stream>>>(agg1, partial);
  k_wsum<<<1, 256, 0, stream>>>(partial, disc_W, wsum);
  k_posneg<<<NNODES / 8, 256, 0, stream>>>(agg1, agg2, wsum, out);
}
